// KA_GNN_two_37142877176048
// MI455X (gfx1250) — matrix operand order repaired and checked by reading, not compiled, not run on hardware
//
#include <hip/hip_runtime.h>
#include <hip/hip_bf16.h>

typedef __attribute__((ext_vector_type(16))) _Float16 v16h;
typedef __attribute__((ext_vector_type(8)))  _Float16 v8h;
typedef __attribute__((ext_vector_type(8)))  float    v8f;

#define HID   64
#define GRD   4
#define FDIM  512   // HID * GRD * 2 (cos/sin)
#define HPAD  68    // padded LDS row stride (floats): bank = (4*row + i) % 64
#define WPAD  520   // padded LDS weight row stride (halfs): lane bank = 4*colL

// ---------------------------------------------------------------------------
// Build transposed f16 weight tables Wt[j][f], f = i*8 + g*2 + t
// (t=0 -> cos coeff = coeffs[0], t=1 -> sin coeff = coeffs[1])
// coeffs layout: [2][HID(out j)][HID(in i)][GRD]
// ---------------------------------------------------------------------------
__global__ void prep_weights(const float* __restrict__ w1,
                             const float* __restrict__ w2,
                             _Float16* __restrict__ wt1,
                             _Float16* __restrict__ wt2) {
  int idx = blockIdx.x * blockDim.x + threadIdx.x;
  if (idx >= 2 * HID * FDIM) return;
  const float* src = (idx < HID * FDIM) ? w1 : w2;
  _Float16*    dst = (idx < HID * FDIM) ? wt1 : wt2;
  int id = idx & (HID * FDIM - 1);
  int j = id >> 9;          // output feature
  int f = id & 511;
  int i = f >> 3;           // input feature
  int g = (f >> 1) & 3;     // frequency index (k-1)
  int t = f & 1;            // 0=cos, 1=sin
  dst[id] = (_Float16)src[((t * HID + j) * HID + i) * GRD + g];
}

__global__ void zero_f32(float* __restrict__ p, int n) {
  int idx = blockIdx.x * blockDim.x + threadIdx.x;
  if (idx < n) p[idx] = 0.f;
}

// ---------------------------------------------------------------------------
// Fused Fourier-KAN layer: y[n,j] = sum_f phi[n,f] * Wt[j,f]
// A fragment (16x32 f16) built per-lane in registers:
//   lanes 0-15 : row M=lane,    K = {0..7} (p=0) and {16..23} (p=1)
//   lanes 16-31: row M=lane-16, K = {8..15}      and {24..31}
// B fragments: bank-conflict-free padded LDS copy of the weight table,
// double-buffered one K-step ahead; loads are issued BEFORE the sincos
// feature block so every ds_load has a full iteration + ~36 VALU of
// independent work before the WMMA that consumes it.
// ---------------------------------------------------------------------------
__global__ __launch_bounds__(256) void kan_layer_wmma(
    const float*    __restrict__ x,   // [Nn, 64]
    const _Float16* __restrict__ wt,  // [64, 512] (j-major, f contiguous)
    float*          __restrict__ y,   // [Nn, 64]
    int Nn) {
  extern __shared__ char smem[];
  _Float16* WL  = (_Float16*)smem;                     // 64 rows * WPAD halfs
  float* Hbase  = (float*)(smem + HID * WPAD * 2);     // 8 waves * 16*HPAD

  // --- cooperative, padded copy of the weight table into LDS (16B chunks) ---
  for (int idx = threadIdx.x; idx < HID * (FDIM / 8); idx += 256) {
    int j  = idx >> 6;          // row
    int cc = idx & 63;          // 16B chunk within row
    v8h v = ((const v8h*)(wt + j * FDIM))[cc];
    *((v8h*)(WL + j * WPAD) + cc) = v;
  }
  __syncthreads();

  const int wave = threadIdx.x >> 5;
  const int lane = threadIdx.x & 31;
  const int tile = blockIdx.x * 8 + wave;
  const int nb   = tile * 16;
  if (nb >= Nn) return;

  float* Ht = Hbase + wave * (16 * HPAD);

  // Stage the 16x64 node tile into LDS with padded rows (coalesced b128)
  for (int idx = lane; idx < 16 * 16; idx += 32) {
    int r = idx >> 4;
    int cc = idx & 15;
    int node = nb + r;
    float4 d = (node < Nn) ? ((const float4*)(x + (size_t)node * HID))[cc]
                           : make_float4(0.f, 0.f, 0.f, 0.f);
    ((float4*)(Ht + r * HPAD))[cc] = d;
  }
  asm volatile("s_wait_dscnt 0" ::: "memory");

  const int row  = lane & 15;
  const int sel  = lane >> 4;   // K-half (A/B), M-half (C/D)
  const int colL = lane & 15;   // N within tile (B, C/D)

  // B-fragment loader: two 16B LDS reads per N-tile, conflict-free
  auto loadB = [&](v16h* bb, int c) {
    #pragma unroll
    for (int nt = 0; nt < 4; ++nt) {
      const v8h* bp =
          (const v8h*)(WL + ((nt * 16 + colL) * WPAD + c * 32 + sel * 8));
      v8h lo = bp[0];
      v8h hi = bp[2];
      #pragma unroll
      for (int t = 0; t < 8; ++t) { bb[nt][t] = lo[t]; bb[nt][t + 8] = hi[t]; }
    }
  };

  // A-fragment builder: one sincos + angle-addition recurrence per input
  auto buildA = [&](int c) -> v16h {
    v16h a;
    #pragma unroll
    for (int p = 0; p < 2; ++p) {
      int i = c * 4 + sel + p * 2;
      float xv = Ht[row * HPAD + i];
      float s1, c1;
      __sincosf(xv, &s1, &c1);
      float ck = c1, sk = s1;
      a[p * 8 + 0] = (_Float16)c1;
      a[p * 8 + 1] = (_Float16)s1;
      #pragma unroll
      for (int k = 1; k < 4; ++k) {
        float cn = ck * c1 - sk * s1;
        float sn = sk * c1 + ck * s1;
        a[p * 8 + 2 * k]     = (_Float16)cn;
        a[p * 8 + 2 * k + 1] = (_Float16)sn;
        ck = cn; sk = sn;
      }
    }
    return a;
  };

  v8f acc[4] = {};
  v16h b0[4], b1[4];
  loadB(b0, 0);

  #pragma unroll
  for (int c = 0; c < 16; ++c) {
    if ((c & 1) == 0) {
      if (c < 15) loadB(b1, c + 1);        // issue next-step loads FIRST
      v16h a = buildA(c);                  // ~36 VALU + 2 trans of cover
      #pragma unroll
      for (int nt = 0; nt < 4; ++nt)
        acc[nt] = __builtin_amdgcn_wmma_f32_16x16x32_f16(
            false, a, false, b0[nt], (short)0, acc[nt], false, false);
    } else {
      if (c < 15) loadB(b0, c + 1);
      v16h a = buildA(c);
      #pragma unroll
      for (int nt = 0; nt < 4; ++nt)
        acc[nt] = __builtin_amdgcn_wmma_f32_16x16x32_f16(
            false, a, false, b1[nt], (short)0, acc[nt], false, false);
    }
  }

  // ---- Store D: lanes 0-15 -> M=v, lanes 16-31 -> M=8+v; N = colL ----
  #pragma unroll
  for (int nt = 0; nt < 4; ++nt) {
    #pragma unroll
    for (int v = 0; v < 8; ++v) {
      int node = nb + sel * 8 + v;
      if (node < Nn)
        y[(size_t)node * HID + nt * 16 + colL] = acc[nt][v];
    }
  }
}

// ---------------------------------------------------------------------------
// Edge scatter: agg[dst[e], :] += m[src[e], :]   (16 lanes per edge, float4)
// ---------------------------------------------------------------------------
__global__ void edge_scatter(const float* __restrict__ m,
                             const int* __restrict__ src,
                             const int* __restrict__ dst,
                             float* __restrict__ agg, int E) {
  int gid = blockIdx.x * blockDim.x + threadIdx.x;
  int e = gid >> 4;
  if (e >= E) return;
  int q = gid & 15;
  int s = src[e];
  int d = dst[e];
  float4 v = ((const float4*)(m + (size_t)s * HID))[q];
  float* ap = agg + (size_t)d * HID + q * 4;
  atomicAdd(ap + 0, v.x);
  atomicAdd(ap + 1, v.y);
  atomicAdd(ap + 2, v.z);
  atomicAdd(ap + 3, v.w);
}

// ---------------------------------------------------------------------------
// leaky_relu(agg + h1) then segment-sum over sorted graph_ids.
// 64 threads = one column each; run-length accumulate, flush on graph change.
// ---------------------------------------------------------------------------
__global__ void act_pool(const float* __restrict__ agg,
                         const float* __restrict__ h1,
                         const int* __restrict__ gids,
                         float* __restrict__ pool, int Nn) {
  int j = threadIdx.x;          // 0..63
  int n0 = blockIdx.x * 256;
  float accv = 0.f;
  int cur = -1;
  for (int t = 0; t < 256; ++t) {
    int n = n0 + t;
    if (n >= Nn) break;
    int g = gids[n];            // uniform -> scalar broadcast
    float v = agg[(size_t)n * HID + j] + h1[(size_t)n * HID + j];
    v = v > 0.f ? v : 0.01f * v;
    if (g != cur) {
      if (cur >= 0) atomicAdd(&pool[(size_t)cur * HID + j], accv);
      accv = 0.f;
      cur = g;
    }
    accv += v;
  }
  if (cur >= 0) atomicAdd(&pool[(size_t)cur * HID + j], accv);
}

// ---------------------------------------------------------------------------
// Readout: sigmoid( sum_j cos(y)*w0[j] + sin(y)*w1[j] + b )   (grid=1, out=1)
// lin1_w layout [2][1][64][1] -> w[j] (cos), w[64+j] (sin)
// ---------------------------------------------------------------------------
__global__ void readout(const float* __restrict__ pool,
                        const float* __restrict__ w,
                        const float* __restrict__ b,
                        float* __restrict__ out, int G) {
  int g = blockIdx.x * blockDim.x + threadIdx.x;
  if (g >= G) return;
  float acc = b[0];
  #pragma unroll 4
  for (int j = 0; j < HID; ++j) {
    float xv = pool[(size_t)g * HID + j];
    float s, c;
    __sincosf(xv, &s, &c);
    acc += c * w[j] + s * w[HID + j];
  }
  out[g] = 1.f / (1.f + __expf(-acc));
}

// ---------------------------------------------------------------------------
extern "C" void kernel_launch(void* const* d_in, const int* in_sizes, int n_in,
                              void* d_out, int out_size, void* d_ws, size_t ws_size,
                              hipStream_t stream) {
  const float* h     = (const float*)d_in[0];
  const float* kanw  = (const float*)d_in[1];
  const float* layw  = (const float*)d_in[2];
  const float* lin1w = (const float*)d_in[3];
  const float* lin1b = (const float*)d_in[4];
  const int*   src   = (const int*)d_in[5];
  const int*   dst   = (const int*)d_in[6];
  const int*   gids  = (const int*)d_in[7];

  const int Nn = in_sizes[0] / HID;   // 100000
  const int E  = in_sizes[5];         // 1.6M
  const int G  = out_size;            // 256 (out=1 per graph)

  // workspace carve-up (~77 MB)
  char* ws = (char*)d_ws;
  _Float16* wt1 = (_Float16*)ws;  ws += (size_t)HID * FDIM * 2;
  _Float16* wt2 = (_Float16*)ws;  ws += (size_t)HID * FDIM * 2;
  float* h1    = (float*)ws;      ws += (size_t)Nn * HID * 4;
  float* mnode = (float*)ws;      ws += (size_t)Nn * HID * 4;
  float* agg   = (float*)ws;      ws += (size_t)Nn * HID * 4;
  float* pool  = (float*)ws;

  prep_weights<<<(2 * HID * FDIM + 255) / 256, 256, 0, stream>>>(kanw, layw, wt1, wt2);
  zero_f32<<<(Nn * HID + 255) / 256, 256, 0, stream>>>(agg, Nn * HID);
  zero_f32<<<(G * HID + 255) / 256, 256, 0, stream>>>(pool, G * HID);

  const int tiles  = (Nn + 15) / 16;
  const int blocks = (tiles + 7) / 8;
  const size_t shmem = (size_t)HID * WPAD * 2 + (size_t)8 * 16 * HPAD * 4; // 101376 B
  kan_layer_wmma<<<blocks, 256, shmem, stream>>>(h, wt1, h1, Nn);
  kan_layer_wmma<<<blocks, 256, shmem, stream>>>(h1, wt2, mnode, Nn);

  long long eth = (long long)E * 16;
  edge_scatter<<<(int)((eth + 255) / 256), 256, 0, stream>>>(mnode, src, dst, agg, E);

  act_pool<<<(Nn + 255) / 256, 64, 0, stream>>>(agg, h1, gids, pool, Nn);
  readout<<<(G + 255) / 256, 256, 0, stream>>>(pool, lin1w, lin1b, (float*)d_out, G);
}
